// SimulatedRingDilatedAttention_86517821216465
// MI455X (gfx1250) — matrix operand order repaired and checked by reading, not compiled, not run on hardware
//
#include <hip/hip_runtime.h>

// MI455X / gfx1250: wave32, WMMA 16x16x32 f16 -> f32 accum.
// Per-position head-mixing attention:
//   T = K @ Q^T  (T = S^T; softmax over g = per-lane reduction + 1 xor-16),
//   P^T(C-layout) == P(A-layout) -> feeds O = P @ V directly, no LDS.
// f32->f16 via v_cvt_pk_rtz_f16_f32 (packed, 2 values/instr).
typedef __attribute__((ext_vector_type(16))) _Float16 v16h;
typedef __attribute__((ext_vector_type(8)))  _Float16 v8h;
typedef __attribute__((ext_vector_type(4)))  _Float16 v4h;
typedef __attribute__((ext_vector_type(2)))  _Float16 v2h;
typedef __attribute__((ext_vector_type(8)))  float    v8f;
typedef __attribute__((ext_vector_type(4)))  float    f32x4;

#define HH 32
#define DD 128
#define HD (HH * DD)                  // 4096 floats per position per tensor
#define CEXP 0.12753785457875646f     // (1/sqrt(128)) * log2(e)
#define WAVES_PER_BLOCK 8

__device__ __forceinline__ f32x4 ldnt4(const float* p) {
    return __builtin_nontemporal_load((const f32x4*)p);
}

__device__ __forceinline__ v2h pk(float x, float y) {
    return (v2h)__builtin_amdgcn_cvt_pkrtz(x, y);   // v_cvt_pk_rtz_f16_f32
}
__device__ __forceinline__ v4h cat2(v2h a, v2h b) {
    return __builtin_shufflevector(a, b, 0, 1, 2, 3);
}
__device__ __forceinline__ v8h cat4(v4h a, v4h b) {
    return __builtin_shufflevector(a, b, 0, 1, 2, 3, 4, 5, 6, 7);
}
__device__ __forceinline__ v16h cat8(v8h a, v8h b) {
    return __builtin_shufflevector(a, b, 0, 1, 2, 3, 4, 5, 6, 7,
                                   8, 9, 10, 11, 12, 13, 14, 15);
}
// 8 contiguous f32 (two float4s) -> 8 packed f16
__device__ __forceinline__ v8h cvt8(f32x4 lo, f32x4 hi) {
    return cat4(cat2(pk(lo.x, lo.y), pk(lo.z, lo.w)),
                cat2(pk(hi.x, hi.y), pk(hi.z, hi.w)));
}

__global__ __launch_bounds__(WAVES_PER_BLOCK * 32)
void srda_headmix_attn(const float* __restrict__ Q,
                       const float* __restrict__ Km,
                       const float* __restrict__ V,
                       float* __restrict__ O,
                       int n_pos)
{
    const int wave = threadIdx.x >> 5;
    const int lane = threadIdx.x & 31;
    const int half = lane >> 4;    // which 16-lane half of the wave
    const int ln   = lane & 15;

    const long pos = (long)blockIdx.x * WAVES_PER_BLOCK + wave;  // b*N + n
    if (pos >= n_pos) return;      // wave-uniform; EXEC stays all-ones below

    const float* q = Q  + pos * HD;
    const float* k = Km + pos * HD;
    const float* v = V  + pos * HD;
    float*       o = O  + pos * HD;

    // ---- T = K @ Q^T : rows = g (from K), cols = h (from Q rows).
    // 2x2 tiles of 16x16, contraction over d = 128 = 4 chunks of 32.
    v8f t[2][2];
    t[0][0] = (v8f){}; t[0][1] = (v8f){};
    t[1][0] = (v8f){}; t[1][1] = (v8f){};

#pragma unroll
    for (int kc = 0; kc < 4; ++kc) {
        const int k0 = kc * 32;
        v16h a[2], b[2];
        // A = K tile (16x32 f16). Lane ln = row g; half selects d sub-ranges
        // {0..7,16..23} (half 0) vs {8..15,24..31} (half 1).
#pragma unroll
        for (int mt = 0; mt < 2; ++mt) {
            const float* src = k + (16 * mt + ln) * DD + k0 + 8 * half;
            a[mt] = cat8(cvt8(ldnt4(src + 0),  ldnt4(src + 4)),
                         cvt8(ldnt4(src + 16), ldnt4(src + 20)));
        }
        // B = Q^T tile (32x16 f16). Lane ln = column h = Q row;
        // half selects d sub-ranges {0..7,16..23} (half 0) vs {8..15,24..31} (half 1).
#pragma unroll
        for (int nt = 0; nt < 2; ++nt) {
            const float* src = q + (16 * nt + ln) * DD + k0 + 8 * half;
            b[nt] = cat8(cvt8(ldnt4(src + 0), ldnt4(src + 4)),
                         cvt8(ldnt4(src + 16), ldnt4(src + 20)));
        }
#pragma unroll
        for (int mt = 0; mt < 2; ++mt)
#pragma unroll
            for (int nt = 0; nt < 2; ++nt)
                t[mt][nt] = __builtin_amdgcn_wmma_f32_16x16x32_f16(
                    false, a[mt], false, b[nt], (short)0, t[mt][nt], false, false);
    }

    // ---- softmax over g (rows of T) for each column h.
    // Lane holds 16 of the 32 g-values of its column (g = 16*mt + 8*half + r);
    // the complementary 16 live in lane^16 -> one xor-16 shuffle per reduction.
#pragma unroll
    for (int nt = 0; nt < 2; ++nt) {
        float mx = t[0][nt][0];
#pragma unroll
        for (int r = 1; r < 8; ++r) mx = fmaxf(mx, t[0][nt][r]);
#pragma unroll
        for (int r = 0; r < 8; ++r) mx = fmaxf(mx, t[1][nt][r]);
        mx = fmaxf(mx, __shfl_xor(mx, 16, 32));

        float sm = 0.0f;
#pragma unroll
        for (int mt = 0; mt < 2; ++mt)
#pragma unroll
            for (int r = 0; r < 8; ++r) {
                float e = exp2f((t[mt][nt][r] - mx) * CEXP);
                t[mt][nt][r] = e;
                sm += e;
            }
        sm += __shfl_xor(sm, 16, 32);
        float inv = 1.0f / sm;
#pragma unroll
        for (int mt = 0; mt < 2; ++mt)
#pragma unroll
            for (int r = 0; r < 8; ++r)
                t[mt][nt][r] *= inv;
    }

    // ---- P^T in C-layout IS P in A-layout: pure per-lane packed converts.
    // pA[mi] element j   <- t[0][mi][j]  (g = j    + 8*half)
    // pA[mi] element j+8 <- t[1][mi][j]  (g = j+16 + 8*half)
    v16h pA[2];
#pragma unroll
    for (int mi = 0; mi < 2; ++mi) {
        v8h lo = cat4(cat2(pk(t[0][mi][0], t[0][mi][1]), pk(t[0][mi][2], t[0][mi][3])),
                      cat2(pk(t[0][mi][4], t[0][mi][5]), pk(t[0][mi][6], t[0][mi][7])));
        v8h hi = cat4(cat2(pk(t[1][mi][0], t[1][mi][1]), pk(t[1][mi][2], t[1][mi][3])),
                      cat2(pk(t[1][mi][4], t[1][mi][5]), pk(t[1][mi][6], t[1][mi][7])));
        pA[mi] = cat8(lo, hi);
    }

    // ---- O = P @ V : 2x8 tiles, single K=32 WMMA step per tile.
#pragma unroll
    for (int ni = 0; ni < 8; ++ni) {
        // B = V tile (32x16). Lane ln = column d; per-lane stride-DD gather,
        // but lanes 0..15 read consecutive d -> each b32 load is two fully
        // coalesced 64B segments (128B per wave instruction).
        const float* vs = v + (8 * half) * DD + 16 * ni + ln;
        float vf[16];
#pragma unroll
        for (int j = 0; j < 16; ++j)
            vf[j] = __builtin_nontemporal_load(vs + (j + (j & 8)) * DD);
        v16h vb = cat8(cat4(cat2(pk(vf[0],  vf[1]),  pk(vf[2],  vf[3])),
                            cat2(pk(vf[4],  vf[5]),  pk(vf[6],  vf[7]))),
                       cat4(cat2(pk(vf[8],  vf[9]),  pk(vf[10], vf[11])),
                            cat2(pk(vf[12], vf[13]), pk(vf[14], vf[15]))));

        v8f o0 = (v8f){};
        v8f o1 = (v8f){};
        o0 = __builtin_amdgcn_wmma_f32_16x16x32_f16(false, pA[0], false, vb,
                                                    (short)0, o0, false, false);
        o1 = __builtin_amdgcn_wmma_f32_16x16x32_f16(false, pA[1], false, vb,
                                                    (short)0, o1, false, false);
#pragma unroll
        for (int r = 0; r < 8; ++r) {
            __builtin_nontemporal_store(o0[r], o + (8 * half + r) * DD + 16 * ni + ln);
            __builtin_nontemporal_store(o1[r], o + (16 + 8 * half + r) * DD + 16 * ni + ln);
        }
    }
}

extern "C" void kernel_launch(void* const* d_in, const int* in_sizes, int n_in,
                              void* d_out, int out_size, void* d_ws, size_t ws_size,
                              hipStream_t stream) {
    (void)n_in; (void)out_size; (void)d_ws; (void)ws_size;
    const float* Q = (const float*)d_in[0];
    const float* K = (const float*)d_in[1];
    const float* V = (const float*)d_in[2];
    float* O = (float*)d_out;

    const int n_pos = in_sizes[0] / HD;  // B*N = 16384
    dim3 block(WAVES_PER_BLOCK * 32);
    dim3 grid((n_pos + WAVES_PER_BLOCK - 1) / WAVES_PER_BLOCK);
    hipLaunchKernelGGL(srda_headmix_attn, grid, block, 0, stream, Q, K, V, O, n_pos);
}
